// ConvOffset2D_1614907703719
// MI455X (gfx1250) — matrix operand order repaired and checked by reading, not compiled, not run on hardware
//
#include <hip/hip_runtime.h>
#include <hip/hip_bf16.h>

typedef __attribute__((ext_vector_type(16))) _Float16 v16h;
typedef __attribute__((ext_vector_type(8)))  float    v8f;

// Problem constants (from reference setup_inputs)
#define BB 32
#define CC 128
#define HH 64
#define WW 64
#define HW (HH * WW)          // 4096
#define XOFF_ELEMS ((long)BB * CC * HH * WW)   // 16,777,216 floats

// One workgroup per (b, h) output row. 128 threads = 4 waves.
// Phase 1: T[p, j] = sum_c x[b, c, rowblk(p), col(p)] * w[c, tap j]
//          via v_wmma_f32_16x16x32_f16  (M=192, N=16 (9 taps), K=128)
// Phase 2: off[w] = sum over valid taps of T
// Phase 3: vertical lerp sampling + coords output
__global__ __launch_bounds__(128) void conv_offset2d_fused(
    const float* __restrict__ xg,    // (B, C, H, W)
    const float* __restrict__ cw,    // (1, C, 3, 3) flat: cw[c*9 + dy*3 + dx]
    float* __restrict__ out)         // [x_offset | coords], both fp32
{
    __shared__ _Float16 wB[CC][16];     // B panel: wB[k][j], j>=9 -> 0   (4 KB)
    __shared__ float    Tl[192][17];    // T tiles, padded stride 17      (~13 KB)
    __shared__ float    offRow[WW];

    const int tid  = threadIdx.x;
    const int wave = tid >> 5;
    const int lane = tid & 31;
    const int hi   = lane >> 4;         // lane half (0/1)
    const int lo   = lane & 15;

    const int wg = blockIdx.x;          // 0 .. B*H-1
    const int b  = wg / HH;
    const int h  = wg % HH;

    // ---- stage weights into LDS as f16 B-panel (each thread owns one k row)
    {
        const int k = tid;              // 0..127
        #pragma unroll
        for (int j = 0; j < 16; ++j)
            wB[k][j] = (j < 9) ? (_Float16)cw[k * 9 + j] : (_Float16)0.0f;
    }
    __syncthreads();

    // ---- Phase 1: 12 M-tiles of 16 positions; wave w owns tiles [3w, 3w+3)
    const long xbatch = (long)b * CC * HW;      // base of this batch in x
    #pragma unroll
    for (int tt = 0; tt < 3; ++tt) {
        const int t = wave * 3 + tt;
        const int p   = t * 16 + lo;            // position within 3-row block
        const int row = p >> 6;                 // 0..2  -> input row h-1+row
        const int col = p & 63;
        const int y   = h - 1 + row;
        const bool valid = (y >= 0) && (y < HH);
        const int ycl = valid ? y : 0;          // clamp so loads stay in-bounds
        const long sbase = xbatch + (long)ycl * WW + col;   // + k*HW per channel

        v8f acc = {};
        #pragma unroll
        for (int ks = 0; ks < 4; ++ks) {
            const int kbase = ks * 32;

            // A tile: 16x32 f16, documented layout (k = e+8*hi | e+8+8*hi)
            v16h a;
            #pragma unroll
            for (int e = 0; e < 16; ++e) {
                const int k = kbase + 8 * hi + ((e < 8) ? e : e + 8);
                const float v = xg[sbase + (long)k * HW];
                a[e] = valid ? (_Float16)v : (_Float16)0.0f;
            }

            // B tile: 32x16 f16 from LDS (N = lane%16, K = e+8*hi | e+8+8*hi)
            v16h bm;
            #pragma unroll
            for (int e = 0; e < 16; ++e)
                bm[e] = wB[kbase + 8 * hi + ((e < 8) ? e : e + 8)][lo];

            acc = __builtin_amdgcn_wmma_f32_16x16x32_f16(
                false, a, false, bm, (short)0, acc, false, false);
        }

        // D layout: element r -> (M = r + 8*hi, N = lane%16)
        #pragma unroll
        for (int r = 0; r < 8; ++r)
            Tl[t * 16 + r + 8 * hi][lo] = acc[r];
    }
    __syncthreads();

    // ---- Phase 2: tap reduction -> offset row (threads 0..63)
    if (tid < WW) {
        const int w = tid;
        float off = 0.0f;
        #pragma unroll
        for (int dy = 0; dy < 3; ++dy) {
            #pragma unroll
            for (int dx = 0; dx < 3; ++dx) {
                const int xc = w + dx - 1;
                if (xc >= 0 && xc < WW)
                    off += Tl[dy * WW + xc][dy * 3 + dx];
            }
        }
        offRow[w] = off;
    }
    __syncthreads();

    // ---- Phase 3: vertical lerp sampling + outputs
    float2* const cb = (float2*)(out + XOFF_ELEMS);
    for (int i = tid; i < CC * WW; i += 128) {
        const int c = i >> 6;
        const int w = i & 63;

        const float yraw = (float)h + offRow[w];
        const float yc   = fminf(fmaxf(yraw, 0.0f), (float)(HH - 1));
        const float y0f  = floorf(yc);
        const float y1f  = ceilf(yc);
        const float ty   = yc - y0f;

        const long chbase = ((long)(b * CC + c)) * HW;
        const float v0 = xg[chbase + (int)y0f * WW + w];
        const float v1 = xg[chbase + (int)y1f * WW + w];
        const float val = v0 + (v1 - v0) * ty;

        const long idx = chbase + h * WW + w;   // same flat index for both outputs
        out[idx] = val;                          // x_offset
        cb[idx]  = make_float2(yc, (float)w);    // coords_clipped (y, x)
    }
}

extern "C" void kernel_launch(void* const* d_in, const int* in_sizes, int n_in,
                              void* d_out, int out_size, void* d_ws, size_t ws_size,
                              hipStream_t stream) {
    const float* x  = (const float*)d_in[0];   // (32,128,64,64) fp32
    const float* cw = (const float*)d_in[1];   // (1,128,3,3)    fp32
    float* out = (float*)d_out;                // [x_offset | coords] fp32

    (void)in_sizes; (void)n_in; (void)out_size; (void)d_ws; (void)ws_size;

    dim3 grid(BB * HH);    // 2048 workgroups: one per (b, h)
    dim3 block(128);       // 4 waves
    conv_offset2d_fused<<<grid, block, 0, stream>>>(x, cw, out);
}
